// PraxisPoly_15994458211120
// MI455X (gfx1250) — matrix operand order repaired and checked by reading, not compiled, not run on hardware
//
#include <hip/hip_runtime.h>
#include <hip/hip_bf16.h>

// ---------------------------------------------------------------------------
// PraxisPoly on MI455X (gfx1250): fused polynomial MLP
//   h   = x @ Wd + bd                       (GEMM  16384x2048x1024)
//   acc = sum_i h^{i+1}*c_i + sum_{i<5} (h^{i+1} @ X_i)*h
//   out = acc @ Wu + bu + x                 (GEMM  16384x1024x2048)
// v_wmma_f32_16x16x32_bf16 GEMMs, double-buffered LDS staged with
// global_load_async_to_lds_b128 (ASYNCcnt). All 8 B fragments + A fragment
// are loaded into distinct registers before the WMMA chain: one s_wait_dscnt
// per k-step, then 8 back-to-back WMMAs. Weights pre-transposed to bf16
// [N][K]; all intermediates (~178MB) sit inside the 192MB L2.
// ---------------------------------------------------------------------------

typedef __attribute__((ext_vector_type(16))) __bf16          v16bf;
typedef __attribute__((ext_vector_type(8)))  float           v8f;
typedef __attribute__((ext_vector_type(8)))  unsigned short  ushort8;

union ABF {            // reinterpret two b128 LDS reads as a WMMA bf16 fragment
    ushort8 h[2];
    v16bf   v;
};

__device__ __forceinline__ unsigned short f2bf(float f) {
    unsigned u = __builtin_bit_cast(unsigned, f);
    unsigned r = u + 0x7FFFu + ((u >> 16) & 1u);   // round-to-nearest-even
    return (unsigned short)(r >> 16);
}
__device__ __forceinline__ float bf2f(unsigned short b) {
    unsigned u = ((unsigned)b) << 16;
    return __builtin_bit_cast(float, u);
}
__device__ __forceinline__ unsigned lds_off(const void* p) {
    // AS3->generic cast keeps the LDS byte offset in the low 32 bits
    return (unsigned)(uintptr_t)p;
}
// Async copy 32B global->LDS (2 x b128); 'offset:' applies to BOTH sides.
__device__ __forceinline__ void async_copy32(unsigned lds, unsigned gbyteoff,
                                             const void* base) {
    asm volatile("global_load_async_to_lds_b128 %0, %1, %2 offset:0\n\t"
                 "global_load_async_to_lds_b128 %0, %1, %2 offset:16"
                 :: "v"(lds), "v"(gbyteoff), "s"(base) : "memory");
}

#define TM 128
#define TN 128
#define TK 32
#define LDS_STRIDE 40   // ushorts; 80B row stride: 16B-aligned b128, spread banks

#define MODE_DOWN  0
#define MODE_CROSS 1
#define MODE_UP    2

#define RRED 1024       // reduced dim (scratch column count)

// ---------------- weight pre-pass: transpose f32 [K][N] -> bf16 [N][K] -----
__global__ __launch_bounds__(256)
void transpose_to_bf16(const float* __restrict__ in,
                       unsigned short* __restrict__ out, int K, int N) {
    __shared__ float tile[32][33];
    const size_t batch = (size_t)K * N;
    const float*     inb  = in  + (size_t)blockIdx.z * batch;
    unsigned short*  outb = out + (size_t)blockIdx.z * batch;
    const int n0 = blockIdx.x * 32, k0 = blockIdx.y * 32;
    const int tx = threadIdx.x & 31, ty = threadIdx.x >> 5;   // 8 rows/pass
    #pragma unroll
    for (int r = ty; r < 32; r += 8)
        tile[r][tx] = inb[(size_t)(k0 + r) * N + (n0 + tx)];
    __syncthreads();
    #pragma unroll
    for (int r = ty; r < 32; r += 8)
        outb[(size_t)(n0 + r) * K + (k0 + tx)] = f2bf(tile[tx][r]);
}

// ---------------- fused GEMM + polynomial epilogue -------------------------
template<int MODE, bool ABF16>
__global__ __launch_bounds__(256)
void gemm_poly(const void*           __restrict__ Aptr,  // f32 (x/acc) or bf16 (xp)
               const unsigned short* __restrict__ BwT,   // bf16 weights, [N][K]
               int K, int N,
               const float* __restrict__ bias,
               const float* __restrict__ coeff,
               unsigned short* __restrict__ hbuf,
               const unsigned short* __restrict__ xp_in,
               unsigned short* __restrict__ xp_out,
               float* __restrict__ accbuf,
               const float* __restrict__ xres,
               float* __restrict__ out)
{
    __shared__ __align__(16) unsigned short As[2][TM][LDS_STRIDE];
    __shared__ __align__(16) unsigned short Bs[2][TN][LDS_STRIDE];

    const int tid   = threadIdx.x;
    const int lane  = tid & 31;
    const int wave  = tid >> 5;               // 0..7
    const int mBase = blockIdx.y * TM;
    const int nBase = blockIdx.x * TN;

    // staging coordinates
    const int sa_m  = tid >> 1;               // 0..127
    const int sa_k0 = (tid & 1) * 16;         // 0 / 16
    const int sb_n  = tid & 127;              // 0..127
    const int sb_k0 = (tid >> 7) * 16;        // 0 / 16

    // fragment coordinates (ISA 7.12.2 layouts, wave32)
    const int fr_row = lane & 15;
    const int akb    = (lane < 16) ? 0 : 8;   // A: K runs {kb..kb+7, kb+16..kb+23}
    const int bkb    = (lane < 16) ? 0 : 8;   // B: K runs {kb..kb+7, kb+16..kb+23}

    v8f acc[8] = {};

    auto stage = [&](int buf, int kk) {
        // ---- B tile: bf16 [N][K] contiguous -> async copy, no VGPR traffic
        {
            unsigned l = lds_off(&Bs[buf][sb_n][sb_k0]);
            unsigned g = (unsigned)(((size_t)(nBase + sb_n) * K + kk + sb_k0) * 2);
            async_copy32(l, g, BwT);
        }
        // ---- A tile
        if (ABF16) {          // xp is bf16 row-major -> async copy
            unsigned l = lds_off(&As[buf][sa_m][sa_k0]);
            unsigned g = (unsigned)(((size_t)(mBase + sa_m) * K + kk + sa_k0) * 2);
            async_copy32(l, g, Aptr);
        } else {              // f32 source: load, convert, ds_store
            const float* ap = (const float*)Aptr + (size_t)(mBase + sa_m) * K + kk + sa_k0;
            float buf16[16];
            *(float4*)&buf16[0]  = *(const float4*)(ap + 0);
            *(float4*)&buf16[4]  = *(const float4*)(ap + 4);
            *(float4*)&buf16[8]  = *(const float4*)(ap + 8);
            *(float4*)&buf16[12] = *(const float4*)(ap + 12);
            #pragma unroll
            for (int j = 0; j < 8; ++j) {
                unsigned lo = f2bf(buf16[2 * j]);
                unsigned hi = f2bf(buf16[2 * j + 1]);
                *(unsigned*)&As[buf][sa_m][sa_k0 + 2 * j] = lo | (hi << 16);
            }
        }
    };

    const int nsteps = K / TK;
    stage(0, 0);                               // prologue fill of buffer 0

    for (int s = 0; s < nsteps; ++s) {
        const int cur = s & 1;
        __syncthreads();                       // compute(s-1) done: buf cur^1 free
        if (s + 1 < nsteps) {
            stage(cur ^ 1, (s + 1) * TK);      // prefetch next tile (overlaps compute)
            if constexpr (ABF16) asm volatile("s_wait_asynccnt 0x4" ::: "memory");
            else                 asm volatile("s_wait_asynccnt 0x2" ::: "memory");
        } else {
            asm volatile("s_wait_asynccnt 0x0" ::: "memory");
        }
        __syncthreads();                       // buf cur fully staged, all waves

        // ---- compute: load A + all 8 B fragments into live registers, ----
        // ---- then 8 back-to-back WMMAs (single ds drain per k-step) ------
        ABF afr;
        afr.h[0] = *(const ushort8*)&As[cur][wave * 16 + fr_row][akb];
        afr.h[1] = *(const ushort8*)&As[cur][wave * 16 + fr_row][akb + 16];
        ABF bfr[8];
        #pragma unroll
        for (int t = 0; t < 8; ++t) {
            bfr[t].h[0] = *(const ushort8*)&Bs[cur][t * 16 + fr_row][bkb];
            bfr[t].h[1] = *(const ushort8*)&Bs[cur][t * 16 + fr_row][bkb + 16];
        }
        #pragma unroll
        for (int t = 0; t < 8; ++t) {
            acc[t] = __builtin_amdgcn_wmma_f32_16x16x32_bf16(
                         false, afr.v, false, bfr[t].v, (short)0, acc[t],
                         false, false);
        }
    }

    // ---- epilogue: C layout  m = j + (lane<16?0:8), n = lane&15 ----------
    const int mw = mBase + wave * 16 + ((lane < 16) ? 0 : 8);
    const int nw = nBase + (lane & 15);
    #pragma unroll
    for (int t = 0; t < 8; ++t) {
        const int n = nw + t * 16;
        #pragma unroll
        for (int j = 0; j < 8; ++j) {
            const int m = mw + j;
            const float cval = acc[t][j];
            if constexpr (MODE == MODE_DOWN) {
                const size_t idx = (size_t)m * RRED + n;
                float h = cval + bias[n];
                unsigned short hb = f2bf(h);
                hbuf[idx]   = hb;               // h (bf16)
                xp_out[idx] = hb;               // x_power = h
                accbuf[idx] = h * coeff[n];     // acc = h * coeffs[0]
            } else if constexpr (MODE == MODE_CROSS) {
                const size_t idx = (size_t)m * RRED + n;
                float hv  = bf2f(hbuf[idx]);
                float xpv = bf2f(xp_in[idx]);
                float av  = accbuf[idx];
                av += cval * hv;                // (xp @ cross[i]) * h
                float xpn = xpv * hv;           // xp <- xp * h
                av += xpn * coeff[n];           // + xp * coeffs[i+1]
                accbuf[idx] = av;
                xp_out[idx] = f2bf(xpn);
            } else { // MODE_UP
                const size_t idx = (size_t)m * N + n;
                out[idx] = cval + bias[n] + xres[idx];
            }
        }
    }
}

extern "C" void kernel_launch(void* const* d_in, const int* in_sizes, int n_in,
                              void* d_out, int out_size, void* d_ws, size_t ws_size,
                              hipStream_t stream) {
    const float* x      = (const float*)d_in[0];   // [4,4096,2048]
    const float* down_w = (const float*)d_in[1];   // [2048,1024]
    const float* down_b = (const float*)d_in[2];   // [1024]
    const float* coeffs = (const float*)d_in[3];   // [6,1024]
    const float* crossw = (const float*)d_in[4];   // [5,1024,1024]
    const float* up_w   = (const float*)d_in[5];   // [1024,2048]
    const float* up_b   = (const float*)d_in[6];   // [2048]
    float*       out    = (float*)d_out;

    const int M = 16384;                 // 4*4096 tokens
    const size_t HN = (size_t)M * RRED;  // elements per scratch plane

    char* ws = (char*)d_ws;
    unsigned short* hbuf = (unsigned short*)(ws);            // 32 MB (bf16)
    unsigned short* xpA  = (unsigned short*)(ws + 2 * HN);   // 32 MB
    unsigned short* xpB  = (unsigned short*)(ws + 4 * HN);   // 32 MB
    float*          accb = (float*)(ws + 6 * HN);            // 64 MB
    unsigned short* WdT  = (unsigned short*)(ws + 10 * HN);              // [1024][2048] 4 MB
    unsigned short* XT   = WdT + (size_t)1024 * 2048;                    // 5x[1024][1024] 10 MB
    unsigned short* WuT  = XT  + (size_t)5 * 1024 * 1024;                // [2048][1024] 4 MB

    dim3 block(256);

    // 0) weight pre-pass: f32 [K][N] -> bf16 [N][K]
    transpose_to_bf16<<<dim3(1024 / 32, 2048 / 32, 1), block, 0, stream>>>(down_w, WdT, 2048, 1024);
    transpose_to_bf16<<<dim3(1024 / 32, 1024 / 32, 5), block, 0, stream>>>(crossw, XT, 1024, 1024);
    transpose_to_bf16<<<dim3(2048 / 32, 1024 / 32, 1), block, 0, stream>>>(up_w, WuT, 1024, 2048);

    dim3 gridR(RRED / TN, M / TM);       // (8, 128)  N=1024 GEMMs
    dim3 gridD(2048 / TN, M / TM);       // (16, 128) N=2048 GEMM

    // 1) down-proj + poly seed
    gemm_poly<MODE_DOWN, false><<<gridR, block, 0, stream>>>(
        x, WdT, 2048, RRED, down_b, coeffs /* coeffs[0] */,
        hbuf, nullptr, xpA, accb, nullptr, nullptr);

    // 2) five cross GEMMs with fused polynomial update (xp ping-pong)
    for (int i = 0; i < 5; ++i) {
        const unsigned short* xin  = (i & 1) ? xpB : xpA;
        unsigned short*       xout = (i & 1) ? xpA : xpB;
        gemm_poly<MODE_CROSS, true><<<gridR, block, 0, stream>>>(
            xin, XT + (size_t)i * RRED * RRED, RRED, RRED,
            nullptr, coeffs + (size_t)(i + 1) * RRED,
            hbuf, xin, xout, accb, nullptr, nullptr);
    }

    // 3) up-proj + bias + residual
    gemm_poly<MODE_UP, false><<<gridD, block, 0, stream>>>(
        accb, WuT, RRED, 2048, up_b, nullptr,
        nullptr, nullptr, nullptr, nullptr, x, out);
}
